// EventPropLinear_87342454932160
// MI455X (gfx1250) — matrix operand order repaired and checked by reading, not compiled, not run on hardware
//
#include <hip/hip_runtime.h>
#include <stdint.h>

typedef __attribute__((ext_vector_type(16))) _Float16 v16h;
typedef __attribute__((ext_vector_type(8)))  float    v8f;
typedef __attribute__((ext_vector_type(8)))  unsigned int v8u;

#define B_SZ    128
#define IN_DIM  784
#define OUT_DIM 512
#define T_STEPS 500
#define KPAD    800   // IN padded to multiple of 32
#define LDSK    808   // padded LDS row stride (f16 elems) to dodge bank conflicts
#define NTILE   64    // OUT columns per workgroup
#define MTILE   32    // time rows per chunk (2 WMMA row-blocks)
#define NCHUNK  16    // 16*32 = 512 >= 499 J rows

// One workgroup = (batch b, 64-wide OUT tile). 4 waves, each owning 16 OUT cols.
// Per chunk: stage 32 timesteps of x (f16) -> WMMA over K=800 (2 M-blocks share
// each B fragment) -> J tile to LDS -> 32 serial LIF steps.
__global__ __launch_bounds__(128)
void snn_eventprop_fused(const float* __restrict__ x,
                         const float* __restrict__ w,
                         float* __restrict__ out)
{
    __shared__ _Float16 As[MTILE * LDSK];   // x tile, [time-row][k], ~51.7 KB
    __shared__ float    Js[MTILE * NTILE];  // J tile, [time-row][col], 8 KB

    const int tid   = threadIdx.x;
    const int lane  = tid & 31;
    const int wave  = tid >> 5;                 // 0..3
    const int b     = blockIdx.y;               // batch index
    const int oblk  = blockIdx.x * NTILE;       // OUT tile base
    const int owave = oblk + wave * 16;

    const float a_m = 1.0f - 0.1f / 20.0f;      // 0.995
    const float b_m = 0.1f / 20.0f;             // 0.005
    const float a_s = 1.0f - 0.1f / 5.0f;       // 0.98

    // persistent LIF state: threads 0..63 own column oblk+tid
    float Vst = 0.0f, Ist = 0.0f;
    if (tid < NTILE) {
        out[((size_t)b * OUT_DIM + (oblk + tid)) * T_STEPS + 0] = 0.0f; // t=0
    }

    const int mlan  = lane & 15;   // row (A/C) or column (B) within 16
    const int khalf = lane >> 4;   // low/high K half of the wave

    for (int tc = 0; tc < NCHUNK; ++tc) {
        const int i0 = tc * MTILE;   // J row base == time index into x

        // ---- stage A tile: As[j][k] = (f16) x[b][k][i0+j], zero padded ----
        for (int k = tid; k < KPAD; k += 128) {
            if (k < IN_DIM) {
                const float* xrow = x + ((size_t)b * IN_DIM + k) * T_STEPS + i0;
                #pragma unroll
                for (int g = 0; g < MTILE / 4; ++g) {
                    float4 vx = make_float4(0.f, 0.f, 0.f, 0.f);
                    if (i0 + 4 * g < T_STEPS)   // 500 % 4 == 0: group fully valid
                        vx = *reinterpret_cast<const float4*>(xrow + 4 * g);
                    As[(4 * g + 0) * LDSK + k] = (_Float16)vx.x;
                    As[(4 * g + 1) * LDSK + k] = (_Float16)vx.y;
                    As[(4 * g + 2) * LDSK + k] = (_Float16)vx.z;
                    As[(4 * g + 3) * LDSK + k] = (_Float16)vx.w;
                }
                // overlap next chunk's HBM traffic with this chunk's WMMA phase
                if (i0 + MTILE < T_STEPS)
                    __builtin_prefetch(xrow + MTILE, 0, 0);
            } else {
                #pragma unroll
                for (int j = 0; j < MTILE; ++j)
                    As[j * LDSK + k] = (_Float16)0.f;
            }
        }
        __syncthreads();

        // ---- WMMA: two 16t x 16o accumulators sharing each B fragment ----
        v8f acc0 = {0.f, 0.f, 0.f, 0.f, 0.f, 0.f, 0.f, 0.f};
        v8f acc1 = {0.f, 0.f, 0.f, 0.f, 0.f, 0.f, 0.f, 0.f};
        const unsigned int* As32 = reinterpret_cast<const unsigned int*>(As);

        for (int kc = 0; kc < KPAD / 32; ++kc) {
            const int kbase = kc * 32;

            // B fragment: lane column n; bfrag[j] = W[n][kb + j], j = 0..15.
            // Address-clamped (never predicated): rows k>=784 of A are zero,
            // so clamped (finite) weights contribute exactly zero.
            const int n  = owave + mlan;
            const int kb = kbase + khalf * 16;
            v16h bfrag;
            #pragma unroll
            for (int q = 0; q < 4; ++q) {
                const int kq = kb + 4 * q;
                const int kcl = (kq <= IN_DIM - 4) ? kq : (IN_DIM - 4);
                const float4 wv =
                    *reinterpret_cast<const float4*>(w + (size_t)n * IN_DIM + kcl);
                bfrag[4 * q + 0] = (_Float16)wv.x;
                bfrag[4 * q + 1] = (_Float16)wv.y;
                bfrag[4 * q + 2] = (_Float16)wv.z;
                bfrag[4 * q + 3] = (_Float16)wv.w;
            }

            // A fragments for row blocks h=0 (t rows 0-15) and h=1 (16-31):
            // v0..7 -> K = khalf*16 + {0,2,...,14} (contiguous, matches B)
            #pragma unroll
            for (int h = 0; h < 2; ++h) {
                const int row = 16 * h + mlan;
                v8u au;
                #pragma unroll
                for (int v = 0; v < 8; ++v) {
                const int kl = khalf * 16 + 2 * v;
                    au[v] = As32[row * (LDSK / 2) + ((kbase + kl) >> 1)];
                }
                v16h afrag = __builtin_bit_cast(v16h, au);
                if (h == 0)
                    acc0 = __builtin_amdgcn_wmma_f32_16x16x32_f16(
                               false, afrag, false, bfrag, (short)0, acc0,
                               false, false);
                else
                    acc1 = __builtin_amdgcn_wmma_f32_16x16x32_f16(
                               false, afrag, false, bfrag, (short)0, acc1,
                               false, false);
            }
        }
        __syncthreads();   // everyone done reading As

        // ---- spill J tiles: C layout = (VGPR r -> M = r + 8*khalf, lane -> N) ----
        #pragma unroll
        for (int r = 0; r < 8; ++r) {
            const int mrow = r + 8 * khalf;
            Js[(mrow +  0) * NTILE + wave * 16 + mlan] = acc0[r];
            Js[(mrow + 16) * NTILE + wave * 16 + mlan] = acc1[r];
        }
        __syncthreads();

        // ---- LIF recurrence: 64 lanes, one OUT column each, 32 serial steps ----
        if (tid < NTILE) {
            float* orow = out + ((size_t)b * OUT_DIM + (oblk + tid)) * T_STEPS;
            #pragma unroll
            for (int mm = 0; mm < MTILE; ++mm) {
                const int i = i0 + mm;                 // J row index
                const float jv = Js[mm * NTILE + tid];
                float Vn = a_m * Vst + b_m * Ist;      // uses old I
                Ist = a_s * Ist + jv;
                float spk = (Vn > 1.0f) ? 1.0f : 0.0f;
                Vst = (1.0f - spk) * Vn;               // reset-to-zero
                if (i <= T_STEPS - 2)                  // emit t = i+1 in [1,499]
                    orow[i + 1] = spk;
            }
        }
        __syncthreads();   // protect Js before next chunk overwrites it
    }
}

extern "C" void kernel_launch(void* const* d_in, const int* in_sizes, int n_in,
                              void* d_out, int out_size, void* d_ws, size_t ws_size,
                              hipStream_t stream) {
    (void)in_sizes; (void)n_in; (void)d_ws; (void)ws_size; (void)out_size;
    const float* x = (const float*)d_in[0];   // (128, 784, 500) f32 spikes
    const float* w = (const float*)d_in[1];   // (512, 784) f32
    float* out = (float*)d_out;               // (128, 512, 500) f32

    dim3 grid(OUT_DIM / NTILE, B_SZ);         // (8, 128)
    dim3 block(128);                          // 4 waves (wave32)
    snn_eventprop_fused<<<grid, block, 0, stream>>>(x, w, out);
}
